// CRF_64716567216843
// MI455X (gfx1250) — compile-verified, matrix operand order checked by reading, not run on hardware
//
#include <hip/hip_runtime.h>
#include <hip/hip_bf16.h>

typedef __attribute__((ext_vector_type(16))) _Float16     v16h;
typedef __attribute__((ext_vector_type(8)))  float        v8f;
typedef __attribute__((ext_vector_type(4)))  unsigned int v4u;
typedef __attribute__((ext_vector_type(8)))  int          v8i;
typedef __attribute__((ext_vector_type(4)))  int          v4i;

constexpr int kT  = 48;   // num tags
constexpr int kTP = 64;   // K padded to 2 x 32 for wmma f16
constexpr int kS  = 512;  // sequence length
constexpr int kB  = 1024; // batch

constexpr float kLOG2E = 1.4426950408889634f;
constexpr float kLN2   = 0.6931471805599453f;

union HU { unsigned int u[8]; v16h h; };

// Raw-hardware transcendentals: safe here (exp args <= 0; log args in
// [~1e-3, ~1e5] -- no denormal/inf fixups needed). Branch-free.
__device__ __forceinline__ float fast_exp(float x) {
  return __builtin_amdgcn_exp2f(x * kLOG2E);
}
__device__ __forceinline__ float fast_log(float x) {
  return __builtin_amdgcn_logf(x) * kLN2;
}
__device__ __forceinline__ void cfence() { asm volatile("" ::: "memory"); }

// ---------------------------------------------------------------------------
// TDM: DMA a 16-row x 48-float tile (row stride kS*kT floats) into LDS.
// D# per CDNA5 ISA : group0 {count=1, lds_addr,
// global_addr[56:0], type=2}; group1 {data_size=4B, tensor_dim0=48,
// tensor_dim1=1M, tile_dim0=48, tile_dim1=16, tensor_dim0_stride=kS*kT}.
// 6-arg builtin variant (clang-23 / therock-10.0 headers).
// ---------------------------------------------------------------------------
__device__ __forceinline__ void tdm_load_tile(const float* gptr, unsigned lds_off) {
  unsigned long long ga = (unsigned long long)(size_t)gptr;
  v4u g0 = { 1u,                                   // count=1 (valid descriptor)
             lds_off,                              // lds_addr (bytes)
             (unsigned)(ga & 0xffffffffu),         // global_addr[31:0]
             ((unsigned)((ga >> 32) & 0x01ffffffu)) | 0x80000000u }; // [56:32]|type=2
  v8i g1 = { 0x00020000,                           // wg_mask=0, data_size=2 (4B)
             (int)(48u << 16),                     // tensor_dim0[15:0]=48
             0,                                    // tensor_dim0 hi / tensor_dim1 lo
             (int)((48u << 16) | 0x10u),           // tile_dim0=48 | tensor_dim1 hi16
             16,                                   // tile_dim1=16, tile_dim2=0
             kS * kT,                              // tensor_dim0_stride = 24576
             0, 0 };
  v4i g2 = { 0, 0, 0, 0 };
  v4i g3 = { 0, 0, 0, 0 };
  v8i g4 = { 0, 0, 0, 0, 0, 0, 0, 0 };
  __builtin_amdgcn_tensor_load_to_lds(g0, g1, g2, g3, g4, 0);
}

// ---------------------------------------------------------------------------
// P[to][from_padded] = exp(transitions[from][to]) as f16, zero pad K=48..63
// ---------------------------------------------------------------------------
__global__ void crf_prep_kernel(const float* __restrict__ trans,
                                _Float16* __restrict__ Pg) {
  int i = blockIdx.x * blockDim.x + threadIdx.x;
  if (i >= kT * kTP) return;
  int to = i / kTP, from = i % kTP;
  float v = (from < kT) ? fast_exp(trans[from * kT + to]) : 0.0f;
  Pg[to * kTP + from] = (_Float16)v;
}

// ---------------------------------------------------------------------------
// Gold-path score per batch element (numerator)
// ---------------------------------------------------------------------------
__global__ void crf_score_kernel(const float* __restrict__ em,
                                 const int* __restrict__ tags,
                                 const unsigned char* __restrict__ mask,
                                 const float* __restrict__ trans,
                                 const float* __restrict__ startT,
                                 const float* __restrict__ endT,
                                 float* __restrict__ score) {
  int b = blockIdx.x * blockDim.x + threadIdx.x;
  if (b >= kB) return;
  const int* tg           = tags + (size_t)b * kS;
  const unsigned char* mk = mask + (size_t)b * kS;
  const float* e          = em   + (size_t)b * kS * kT;

  int msum = 0;
  for (int s = 0; s < kS; ++s) msum += (mk[s] != 0);
  int last = msum - 1;

  float sc = startT[tg[0]];
  int prev = tg[0];
  float em_last = 0.0f;
  int   tag_last = 0;
  for (int s = 0; s < kS; ++s) {
    int t = tg[s];
    float et = e[s * kT + t];
    if (s < kS - 1 && mk[s]) sc += et;
    if (s >= 1     && mk[s]) sc += trans[prev * kT + t];
    if (s == last) { em_last = et; tag_last = t; }
    prev = t;
  }
  sc += em_last + endT[tag_last];
  score[b] = sc;
}

// ---------------------------------------------------------------------------
// Forward scan: one wave per 16-batch tile; alpha kept in WMMA C-layout.
// alpha'[to] = log( exp(alpha-m) @ exp(trans) ) + m + e[to]
// Emissions arrive via TDM (double-buffered), matmul via v_wmma f16->f32.
// ---------------------------------------------------------------------------
__global__ __launch_bounds__(32) void crf_scan_kernel(
    const float* __restrict__ em, const unsigned char* __restrict__ mask,
    const float* __restrict__ startT, const float* __restrict__ endT,
    const _Float16* __restrict__ Pg, float* __restrict__ logz) {
  __shared__ __align__(16) _Float16 vA[16 * kTP];    // [m][from_padded] f16
  __shared__ __align__(16) float    emS[2][16 * kT]; // double-buffered TDM dest

  const int lane = threadIdx.x & 31;
  const int nl   = lane & 15;     // position within 16-lane half
  const int hf   = lane >> 4;     // 0 or 1
  const int b0   = blockIdx.x * 16;

  // zero vA (covers padded K=48..63; live region rewritten every step)
  {
    unsigned int* z = reinterpret_cast<unsigned int*>(vA);
#pragma unroll
    for (int i = 0; i < (16 * kTP / 2) / 32; ++i) z[lane + i * 32] = 0u;
  }

  // B fragments (exp(trans)), resident in VGPRs for the whole scan.
  v16h bf[2][3];
#pragma unroll
  for (int kb = 0; kb < 2; ++kb)
#pragma unroll
    for (int nt = 0; nt < 3; ++nt) {
      const unsigned int* src =
          reinterpret_cast<const unsigned int*>(Pg + (nt * 16 + nl) * kTP);
      HU uu;
#pragma unroll
      for (int j = 0; j < 8; ++j) {
        int k = kb * 32 + (hf ? 8 : 0) + (j < 4 ? 2 * j : 16 + 2 * (j - 4));
        uu.u[j] = src[k >> 1];
      }
      bf[kb][nt] = uu.h;
    }

  float sT[3], eTv[3];
#pragma unroll
  for (int nt = 0; nt < 3; ++nt) {
    sT[nt]  = startT[nt * 16 + nl];
    eTv[nt] = endT[nt * 16 + nl];
  }

  const unsigned ldsA[2] = { (unsigned)(size_t)(void*)&emS[0][0],
                             (unsigned)(size_t)(void*)&emS[1][0] };
  const float* base = em + (size_t)b0 * kS * kT;   // tile for step s: base + s*kT

  // Prologue: DMA step 0 and step 1, wait for step 0.
  cfence();
  tdm_load_tile(base + 0 * kT, ldsA[0]);
  tdm_load_tile(base + 1 * kT, ldsA[1]);
  cfence();
  __builtin_amdgcn_s_wait_tensorcnt(1);
  cfence();

  // alpha init: startT + em[:,0,:]
  float alpha[3][8];   // element r -> batch row m = r + 8*hf, col = nt*16+nl
#pragma unroll
  for (int nt = 0; nt < 3; ++nt)
#pragma unroll
    for (int r = 0; r < 8; ++r)
      alpha[nt][r] = sT[nt] + emS[0][(r + 8 * hf) * kT + nt * 16 + nl];

  for (int s = 1; s < kS; ++s) {
    const float* cur = &emS[s & 1][0];

    // previous buffer's ds_loads must be retired before TDM overwrites it
    asm volatile("s_wait_dscnt 0" ::: "memory");
    if (s + 1 < kS) tdm_load_tile(base + (size_t)(s + 1) * kT, ldsA[(s + 1) & 1]);
    cfence();

    // per-row max of alpha (reduce within each 16-lane half)
    float mrow[8];
#pragma unroll
    for (int r = 0; r < 8; ++r)
      mrow[r] = fmaxf(fmaxf(alpha[0][r], alpha[1][r]), alpha[2][r]);
#pragma unroll
    for (int off = 1; off <= 8; off <<= 1)
#pragma unroll
      for (int r = 0; r < 8; ++r)
        mrow[r] = fmaxf(mrow[r], __shfl_xor(mrow[r], off, 32));

    // v = exp(alpha - m) -> f16 into LDS at [m][from]
#pragma unroll
    for (int nt = 0; nt < 3; ++nt)
#pragma unroll
      for (int r = 0; r < 8; ++r)
        vA[(r + 8 * hf) * kTP + nt * 16 + nl] =
            (_Float16)fast_exp(alpha[nt][r] - mrow[r]);
    asm volatile("s_wait_dscnt 0" ::: "memory");

    // A fragments: 16-bit A 16x32 layout, row M = nl
    v16h af[2];
#pragma unroll
    for (int kb = 0; kb < 2; ++kb) {
      HU uu;
#pragma unroll
      for (int j = 0; j < 8; ++j) {
        int k = kb * 32 + (hf ? 8 : 0) + (j < 4 ? 2 * j : 16 + 2 * (j - 4));
        uu.u[j] = *reinterpret_cast<const unsigned int*>(vA + nl * kTP + k);
      }
      af[kb] = uu.h;
    }

    // acc = exp(alpha-m) @ exp(trans) : 6 WMMAs per step
    v8f acc[3];
#pragma unroll
    for (int nt = 0; nt < 3; ++nt) {
      v8f c = {};
      c = __builtin_amdgcn_wmma_f32_16x16x32_f16(false, af[0], false, bf[0][nt],
                                                 (short)0, c, false, false);
      c = __builtin_amdgcn_wmma_f32_16x16x32_f16(false, af[1], false, bf[1][nt],
                                                 (short)0, c, false, false);
      acc[nt] = c;
    }

    // mask for this step: one ballot instead of 8 bpermutes
    unsigned char mbv = mask[(size_t)(b0 + nl) * kS + s];
    unsigned bal = __builtin_amdgcn_ballot_w32(mbv != 0);

    // current emission tile landed?
    cfence();
    if (s + 1 < kS) __builtin_amdgcn_s_wait_tensorcnt(1);
    else            __builtin_amdgcn_s_wait_tensorcnt(0);
    cfence();

#pragma unroll
    for (int r = 0; r < 8; ++r) {
      int  m  = r + 8 * hf;
      bool mm = (bal >> m) & 1u;
#pragma unroll
      for (int nt = 0; nt < 3; ++nt) {
        float an = fast_log(acc[nt][r]) + mrow[r] + cur[m * kT + nt * 16 + nl];
        // Opaque pass-through: forces 'an' to be computed unconditionally so
        // the ternary below lowers to v_cndmask instead of saveexec branches.
        asm volatile("" : "+v"(an));
        alpha[nt][r] = mm ? an : alpha[nt][r];
      }
    }
  }

  // log_z[b] = logsumexp(alpha + endT)
  float mrow[8], ssum[8];
#pragma unroll
  for (int r = 0; r < 8; ++r)
    mrow[r] = fmaxf(fmaxf(alpha[0][r] + eTv[0], alpha[1][r] + eTv[1]),
                    alpha[2][r] + eTv[2]);
#pragma unroll
  for (int off = 1; off <= 8; off <<= 1)
#pragma unroll
    for (int r = 0; r < 8; ++r)
      mrow[r] = fmaxf(mrow[r], __shfl_xor(mrow[r], off, 32));
#pragma unroll
  for (int r = 0; r < 8; ++r)
    ssum[r] = fast_exp(alpha[0][r] + eTv[0] - mrow[r]) +
              fast_exp(alpha[1][r] + eTv[1] - mrow[r]) +
              fast_exp(alpha[2][r] + eTv[2] - mrow[r]);
#pragma unroll
  for (int off = 1; off <= 8; off <<= 1)
#pragma unroll
    for (int r = 0; r < 8; ++r)
      ssum[r] += __shfl_xor(ssum[r], off, 32);

  if (nl == 0) {
#pragma unroll
    for (int r = 0; r < 8; ++r)
      logz[b0 + 8 * hf + r] = fast_log(ssum[r]) + mrow[r];
  }
}

// ---------------------------------------------------------------------------
// out = mean(logz - score)  ( = -mean(llh) )
// ---------------------------------------------------------------------------
__global__ void crf_reduce_kernel(const float* __restrict__ score,
                                  const float* __restrict__ logz,
                                  float* __restrict__ out) {
  __shared__ float sm[256];
  int tid = threadIdx.x;
  float acc = 0.0f;
  for (int i = tid; i < kB; i += 256) acc += logz[i] - score[i];
  sm[tid] = acc;
  __syncthreads();
  for (int w = 128; w > 0; w >>= 1) {
    if (tid < w) sm[tid] += sm[tid + w];
    __syncthreads();
  }
  if (tid == 0) out[0] = sm[0] / (float)kB;
}

extern "C" void kernel_launch(void* const* d_in, const int* in_sizes, int n_in,
                              void* d_out, int out_size, void* d_ws, size_t ws_size,
                              hipStream_t stream) {
  const float*         em     = (const float*)d_in[0];
  const int*           tags   = (const int*)d_in[1];
  const unsigned char* mask   = (const unsigned char*)d_in[2];   // jnp bool: 1 byte
  const float*         trans  = (const float*)d_in[3];
  const float*         startT = (const float*)d_in[4];
  const float*         endT   = (const float*)d_in[5];
  float* out = (float*)d_out;

  char* ws = (char*)d_ws;
  _Float16* Pg    = (_Float16*)ws;              // 48*64*2  = 6 KB
  float*    score = (float*)(ws + 8192);        // 4 KB
  float*    logz  = (float*)(ws + 8192 + 4096); // 4 KB

  crf_prep_kernel<<<(kT * kTP + 255) / 256, 256, 0, stream>>>(trans, Pg);
  crf_score_kernel<<<(kB + 127) / 128, 128, 0, stream>>>(em, tags, mask, trans,
                                                         startT, endT, score);
  crf_scan_kernel<<<kB / 16, 32, 0, stream>>>(em, mask, startT, endT, Pg, logz);
  crf_reduce_kernel<<<1, 256, 0, stream>>>(score, logz, out);
}
